// E2E_50629074486007
// MI455X (gfx1250) — matrix operand order repaired and checked by reading, not compiled, not run on hardware
//
#include <hip/hip_runtime.h>
#include <hip/hip_bf16.h>
#include <math.h>

typedef __bf16 bf16_t;
typedef __bf16 v16bf __attribute__((ext_vector_type(16)));
typedef __bf16 v8bf  __attribute__((ext_vector_type(8)));
typedef float  v8f   __attribute__((ext_vector_type(8)));
typedef unsigned int v4u __attribute__((ext_vector_type(4)));
typedef int v8i __attribute__((ext_vector_type(8)));
typedef int v4i __attribute__((ext_vector_type(4)));

#define B_   64
#define T_   512
#define D_   1024
#define H_   1024
#define G4H  4096
#define BT   (B_ * T_)
#define NBLK 64            // persistent blocks (one 16-wide hidden tile each)

// ---- workspace layout (all offsets 256B-aligned) ----
static constexpr size_t OFF_XBF  = 0;                              // bf16 [BT][D]
static constexpr size_t OFF_WIH  = OFF_XBF + (size_t)BT * D_ * 2;  // bf16 [4H][D]
static constexpr size_t OFF_WHH  = OFF_WIH + (size_t)G4H * D_ * 2; // bf16 [4H][H]
static constexpr size_t OFF_BIAS = OFF_WHH + (size_t)G4H * H_ * 2; // f32  [4H]
static constexpr size_t OFF_XG   = OFF_BIAS + (size_t)G4H * 4;     // bf16 [T][B][4H]
static constexpr size_t OFF_H0   = OFF_XG + (size_t)BT * G4H * 2;  // bf16 [B][H]
static constexpr size_t OFF_H1   = OFF_H0 + (size_t)B_ * H_ * 2;   // bf16 [B][H]
static constexpr size_t OFF_HF   = OFF_H1 + (size_t)B_ * H_ * 2;   // f32  [B][H]
static constexpr size_t OFF_CTR  = OFF_HF + (size_t)B_ * H_ * 4;   // u32 barrier counter

__global__ void k_cvt_bf16(const float* __restrict__ src, bf16_t* __restrict__ dst, int n) {
    int i = blockIdx.x * blockDim.x + threadIdx.x;
    if (i < n) dst[i] = (bf16_t)src[i];
}

__global__ void k_bias_init(const float* __restrict__ bi, const float* __restrict__ bh,
                            float* __restrict__ bias, bf16_t* __restrict__ h0,
                            unsigned* __restrict__ ctr) {
    int i = blockIdx.x * blockDim.x + threadIdx.x;
    if (i < G4H) bias[i] = bi[i] + bh[i];
    if (i < B_ * H_) h0[i] = (bf16_t)0.0f;
    if (i == 0) *ctr = 0u;
}

// Phase 1: xg[t*64+b][4H] = x[b*512+t][:] @ w_ih^T + (b_ih+b_hh), bf16 WMMA, f32 acc.
__global__ void __launch_bounds__(256)
k_gemm_xg(const bf16_t* __restrict__ xbf, const bf16_t* __restrict__ wih,
          const float* __restrict__ bias, bf16_t* __restrict__ xg) {
    const int lane = threadIdx.x & 31;
    const int wave = threadIdx.x >> 5;
    const int ln = lane & 15;
    const int hi = lane >> 4;
    const int m0 = blockIdx.y * 32 + (wave >> 2) * 16;   // row of x (bt index)
    const int n0 = blockIdx.x * 256 + (wave & 3) * 64;   // gate-channel column

    v8f acc[4];
#pragma unroll
    for (int j = 0; j < 4; ++j) {
        float bv = bias[n0 + j * 16 + ln];
#pragma unroll
        for (int r = 0; r < 8; ++r) acc[j][r] = bv;
    }

    const bf16_t* arow = xbf + (size_t)(m0 + ln) * D_;
#pragma unroll 4
    for (int k0 = 0; k0 < D_; k0 += 32) {
        v16bf a;  // A 16x32: low lanes K{0..15}, high lanes K{16..31}
        ((v8bf*)&a)[0] = *(const v8bf*)(arow + k0 + hi * 16);
        ((v8bf*)&a)[1] = *(const v8bf*)(arow + k0 + hi * 16 + 8);
#pragma unroll
        for (int j = 0; j < 4; ++j) {
            const bf16_t* brow = wih + (size_t)(n0 + j * 16 + ln) * D_ + k0 + hi * 16;
            v16bf b = *(const v16bf*)brow; // B 32x16: 16 contiguous K per lane
            acc[j] = __builtin_amdgcn_wmma_f32_16x16x32_bf16(
                false, a, false, b, (short)0, acc[j], false, false);
        }
    }

#pragma unroll
    for (int j = 0; j < 4; ++j) {
        int n = n0 + j * 16 + ln;
#pragma unroll
        for (int r = 0; r < 8; ++r) {
            int m = m0 + r + hi * 8;           // bt = b*512 + t
            int b = m >> 9, t = m & 511;
            size_t idx = (size_t)(t * 64 + b) * G4H + n;
            __builtin_nontemporal_store((bf16_t)acc[j][r], xg + idx);
        }
    }
}

// Phase 2: persistent LSTM recurrence.
// 64 resident blocks x 4 waves. Block owns one 16-wide hidden tile; wave owns one
// 16(batch)x16(hidden) tile of all 4 gates. w_hh slice (128KB) staged once into LDS
// via the Tensor Data Mover; cell state c lives in registers across all 512 steps;
// steps separated by a device-scope atomic grid barrier.
__global__ void __launch_bounds__(128)
k_lstm_persistent(const bf16_t* __restrict__ xg, const bf16_t* __restrict__ whh,
                  bf16_t* __restrict__ h0, bf16_t* __restrict__ h1,
                  float* __restrict__ hf, unsigned* __restrict__ ctr) {
    extern __shared__ bf16_t lwhh[];   // [4 gates][16 rows][1024 K] bf16 = 128 KB
    const int lane = threadIdx.x & 31;
    const int wave = threadIdx.x >> 5;
    const int ln = lane & 15;
    const int hi = lane >> 4;
    const int m0 = wave * 16;                 // batch tile of this wave
    const int n0 = blockIdx.x * 16;           // hidden tile of this block
    const int n  = n0 + ln;

    // ---- stage w_hh[g*H + n0 .. +16][0..H) into LDS, one TDM op per wave (gate) ----
#if __has_builtin(__builtin_amdgcn_tensor_load_to_lds)
    {
        const int g = wave;
        unsigned lds_off = (unsigned)(uintptr_t)&lwhh[(size_t)g * 16 * H_];
        unsigned long long ga =
            (unsigned long long)(uintptr_t)(whh + ((size_t)g * H_ + n0) * H_);
        v4u g0 = { 1u,                                   // count=1, user mode
                   lds_off,                              // lds_addr
                   (unsigned)(ga & 0xffffffffu),         // global_addr[31:0]
                   (unsigned)((ga >> 32) & 0x01ffffffu) | 0x80000000u }; // addr[56:32] | type=2
        v8i g1 = { 0x00010000,   // data_size = 2 bytes
                   0x04000000,   // tensor_dim0 = 1024  (bits[79:48], low half here)
                   0x10000000,   // tensor_dim1 = 4096  (bits[111:80], low half here)
                   0x04000000,   // tile_dim0 = 1024    (bits[127:112])
                   16,           // tile_dim1 = 16 rows
                   1024,         // tensor_dim0_stride = 1024 elements
                   0, 0 };
        v4i gz4 = { 0, 0, 0, 0 };          // groups 2/3 unused (2D tile)
        v8i gz8 = { 0, 0, 0, 0, 0, 0, 0, 0 };
        __builtin_amdgcn_tensor_load_to_lds(g0, g1, gz4, gz4, gz8, 0);
        __builtin_amdgcn_s_wait_tensorcnt(0);
    }
#else
    {
        // fallback: cooperative global->LDS copy, 16B chunks
        for (int ci = threadIdx.x; ci < 4 * 16 * (H_ / 8); ci += 128) {
            int g = ci / (16 * (H_ / 8));
            int rem = ci % (16 * (H_ / 8));
            int r = rem / (H_ / 8);
            int kc = rem % (H_ / 8);
            *(v8bf*)(lwhh + (size_t)ci * 8) =
                *(const v8bf*)(whh + ((size_t)(g * H_ + n0 + r) * H_) + (size_t)kc * 8);
        }
    }
#endif
    __syncthreads();

    v8f creg;                                 // cell state tile: registers for all T
#pragma unroll
    for (int r = 0; r < 8; ++r) creg[r] = 0.0f;

    bf16_t* hbuf[2] = { h0, h1 };

    for (int t = 0; t < T_; ++t) {
        const bf16_t* hprev = hbuf[t & 1];
        bf16_t* hnext = hbuf[(t + 1) & 1];
        const bf16_t* xgt = xg + (size_t)t * (B_ * G4H);

        v8f acc[4];
#pragma unroll
        for (int g = 0; g < 4; ++g)
#pragma unroll
            for (int r = 0; r < 8; ++r) {
                int m = m0 + r + hi * 8;
                acc[g][r] = (float)xgt[(size_t)m * G4H + g * H_ + n];
            }

        // prefetch next timestep's xg tile while we compute this one
        if (t + 1 < T_) {
            const bf16_t* xgn = xgt + (size_t)B_ * G4H;
#pragma unroll
            for (int g = 0; g < 4; ++g)
                __builtin_prefetch(&xgn[(size_t)(m0 + ln) * G4H + g * H_ + n], 0, 0);
        }

        const bf16_t* arow = hprev + (size_t)(m0 + ln) * H_;
#pragma unroll 4
        for (int k0 = 0; k0 < H_; k0 += 32) {
            v16bf a;
            ((v8bf*)&a)[0] = *(const v8bf*)(arow + k0 + hi * 16);
            ((v8bf*)&a)[1] = *(const v8bf*)(arow + k0 + hi * 16 + 8);
#pragma unroll
            for (int g = 0; g < 4; ++g) {
                const bf16_t* brow =
                    lwhh + (size_t)g * 16 * H_ + (size_t)ln * H_ + k0 + hi * 16;
                v16bf b = *(const v16bf*)brow;       // ds_load from LDS
                acc[g] = __builtin_amdgcn_wmma_f32_16x16x32_bf16(
                    false, a, false, b, (short)0, acc[g], false, false);
            }
        }

#pragma unroll
        for (int r = 0; r < 8; ++r) {
            int m = m0 + r + hi * 8;
            float iv = 1.0f / (1.0f + __expf(-acc[0][r]));
            float fv = 1.0f / (1.0f + __expf(-acc[1][r]));
            float gv = tanhf(acc[2][r]);
            float ov = 1.0f / (1.0f + __expf(-acc[3][r]));
            float cn = fv * creg[r] + iv * gv;
            creg[r] = cn;
            float hn = ov * tanhf(cn);
            hnext[(size_t)m * H_ + n] = (bf16_t)hn;
            if (t == T_ - 1) hf[(size_t)m * H_ + n] = hn;
        }

        // ---- device-wide step barrier (monotonic counter) ----
        __syncthreads();
        __threadfence();                               // release h writes to device scope
        if (threadIdx.x == 0) {
            atomicAdd(ctr, 1u);
            unsigned target = (unsigned)(NBLK * (t + 1));
            while (__hip_atomic_load(ctr, __ATOMIC_RELAXED, __HIP_MEMORY_SCOPE_AGENT) < target)
                __builtin_amdgcn_s_sleep(1);
        }
        __threadfence();                               // acquire other blocks' h writes
        __syncthreads();
    }
}

// Phase 3: gender [64,2] then age [64,7] heads from final h (f32).
__global__ void k_heads(const float* __restrict__ hf, const float* __restrict__ wg,
                        const float* __restrict__ bg, const float* __restrict__ wa,
                        const float* __restrict__ ba, float* __restrict__ out) {
    int tid = blockIdx.x * blockDim.x + threadIdx.x;
    if (tid >= B_ * 9) return;
    int b = tid / 9, j = tid % 9;
    const float* hrow = hf + (size_t)b * H_;
    const float* w;
    float s;
    if (j < 2) { w = wg + (size_t)j * H_;        s = bg[j]; }
    else       { w = wa + (size_t)(j - 2) * H_;  s = ba[j - 2]; }
    for (int k = 0; k < H_; k += 4)
        s += hrow[k] * w[k] + hrow[k+1] * w[k+1] + hrow[k+2] * w[k+2] + hrow[k+3] * w[k+3];
    if (j < 2) out[b * 2 + j] = s;
    else       out[128 + b * 7 + (j - 2)] = s;
}

extern "C" void kernel_launch(void* const* d_in, const int* in_sizes, int n_in,
                              void* d_out, int out_size, void* d_ws, size_t ws_size,
                              hipStream_t stream) {
    const float* x    = (const float*)d_in[0];
    const float* w_ih = (const float*)d_in[1];
    const float* w_hh = (const float*)d_in[2];
    const float* b_ih = (const float*)d_in[3];
    const float* b_hh = (const float*)d_in[4];
    const float* w_g  = (const float*)d_in[5];
    const float* b_g  = (const float*)d_in[6];
    const float* w_a  = (const float*)d_in[7];
    const float* b_a  = (const float*)d_in[8];
    float* out = (float*)d_out;

    char* ws = (char*)d_ws;
    bf16_t* xbf  = (bf16_t*)(ws + OFF_XBF);
    bf16_t* wih  = (bf16_t*)(ws + OFF_WIH);
    bf16_t* whh  = (bf16_t*)(ws + OFF_WHH);
    float*  bias = (float*)(ws + OFF_BIAS);
    bf16_t* xg   = (bf16_t*)(ws + OFF_XG);
    bf16_t* h0   = (bf16_t*)(ws + OFF_H0);
    bf16_t* h1   = (bf16_t*)(ws + OFF_H1);
    float*  hf   = (float*)(ws + OFF_HF);
    unsigned* ctr = (unsigned*)(ws + OFF_CTR);

    // precision conversion + init
    k_cvt_bf16<<<(BT * D_ + 255) / 256, 256, 0, stream>>>(x, xbf, BT * D_);
    k_cvt_bf16<<<(G4H * D_ + 255) / 256, 256, 0, stream>>>(w_ih, wih, G4H * D_);
    k_cvt_bf16<<<(G4H * H_ + 255) / 256, 256, 0, stream>>>(w_hh, whh, G4H * H_);
    k_bias_init<<<(B_ * H_ + 255) / 256, 256, 0, stream>>>(b_ih, b_hh, bias, h0, ctr);

    // phase 1: input projection GEMM
    k_gemm_xg<<<dim3(G4H / 256, BT / 32), 256, 0, stream>>>(xbf, wih, bias, xg);

    // phase 2: persistent recurrence (single launch, 128KB dynamic LDS per block)
    k_lstm_persistent<<<NBLK, 128, 4 * 16 * H_ * sizeof(bf16_t), stream>>>(
        xg, whh, h0, h1, hf, ctr);

    // phase 3: heads
    k_heads<<<(B_ * 9 + 127) / 128, 128, 0, stream>>>(hf, w_g, b_g, w_a, b_a, out);
    (void)in_sizes; (void)n_in; (void)out_size; (void)ws_size;
}
